// ModulatedMambaStack_11725260718318
// MI455X (gfx1250) — compile-verified, matrix operand order checked by reading, not run on hardware
//
#include <hip/hip_runtime.h>

// ---------------- problem constants ----------------
#define DM      768
#define DCOND_  128
#define NL      6
#define DS      16
#define DCONV_  4
#define DI      1536          // EXPAND * D_MODEL
#define DTR     48            // dt_rank
#define DBCN    80            // DTR + 2*DS
#define BATCH_  2
#define SEQ_    1024
#define TTOK    (BATCH_*SEQ_) // 2048 tokens
#define EPS_    1e-5f

// ---------------- CDNA5 WMMA types ----------------
typedef __attribute__((ext_vector_type(16))) __bf16 v16bf;
typedef __attribute__((ext_vector_type(8)))  float  v8f;

__device__ inline unsigned short f32_to_bf16(float f) {
  union { float f; unsigned int u; } c; c.f = f;
  unsigned int u = c.u;
  u += 0x7fffu + ((u >> 16) & 1u);   // round-to-nearest-even
  return (unsigned short)(u >> 16);
}

// low 32 bits of a flat shared-aperture address == LDS byte offset (ISA 10.2)
__device__ inline unsigned lds_off_u32(const void* p) {
  return (unsigned)(uintptr_t)p;
}

// async direct-to-LDS 16B copy, tracked by ASYNCcnt (inline asm: no probe-
// confirmed builtin for GLOBAL_LOAD_ASYNC_TO_LDS_* on this toolchain)
__device__ inline void async_copy_b128(unsigned ldsAddr, const void* gptr) {
  asm volatile("global_load_async_to_lds_b128 %0, %1, off"
               :: "v"(ldsAddr), "v"(gptr) : "memory");
}
__device__ inline void wait_asynccnt0() {
  asm volatile("s_wait_asynccnt 0x0" ::: "memory");
}

// ---------------- converters ----------------
__global__ void k_cvt_bf16(const float* __restrict__ src,
                           unsigned short* __restrict__ dst, int n) {
  int i = blockIdx.x * blockDim.x + threadIdx.x;
  if (i < n) dst[i] = f32_to_bf16(src[i]);
}

__global__ void k_cvt_bf16_2d(const float* __restrict__ src,
                              unsigned short* __restrict__ dst,
                              int rows, int cols, int src_ld) {
  int i = blockIdx.x * blockDim.x + threadIdx.x;
  if (i < rows * cols) {
    int r = i / cols, c = i - r * cols;
    dst[i] = f32_to_bf16(src[(size_t)r * src_ld + c]);
  }
}

// ---------------- WMMA GEMM: C[M,N] = A[M,K] * Bw[N,K]^T ----------------
// A, Bw bf16 row-major (K contiguous); C fp32 row-major.
// Double-buffered LDS staging via GLOBAL_LOAD_ASYNC_TO_LDS_B128 (ASYNCcnt).
// EPI: 0 = none, 1 = +bias[n], 2 = softplus(x + bias[n])
#define BM 128
#define BN 64
#define BK 32
#define LDSB (BK + 8)   // bf16 units, padded against bank conflicts

union FragU { uint4 u[2]; v16bf v; };

template<int EPI>
__global__ void __launch_bounds__(256) k_gemm_wmma(
    const unsigned short* __restrict__ A,
    const unsigned short* __restrict__ Bw,
    const float* __restrict__ bias,
    float* __restrict__ C, int M, int N, int K)
{
  __shared__ __align__(16) unsigned short sA[2][BM * LDSB];
  __shared__ __align__(16) unsigned short sB[2][BN * LDSB];

  const int tid  = threadIdx.x;
  const int lane = tid & 31;
  const int wave = tid >> 5;        // 8 wave32s
  const int wm   = wave >> 1;       // 0..3 -> 32-row strip
  const int wn   = wave & 1;        // 0..1 -> 32-col strip
  const int m0   = blockIdx.y * BM;
  const int n0   = blockIdx.x * BN;

  // fragment addressing per CDNA5 16-bit A/B layout:
  // lanes 0-15: row=lane,   K chunks [0..7] and [16..23]
  // lanes16-31: row=lane-16,K chunks [8..15] and [24..31]
  const int frow = lane & 15;
  const int fkb  = (lane >> 4) << 3;     // 0 or 8 (bf16 units)

  const v8f vz = {0.f,0.f,0.f,0.f,0.f,0.f,0.f,0.f};
  v8f acc[2][2];
  acc[0][0] = vz; acc[0][1] = vz; acc[1][0] = vz; acc[1][1] = vz;

  // staging: 16B (8 bf16) chunks; A tile 128x32 = 512 chunks (2/thread),
  // B tile 64x32 = 256 chunks (1/thread). chunk -> row = cid>>2, col = (cid&3)*8
  auto stageA = [&](int buf, int k0, int cid) {
    int r = cid >> 2, c = (cid & 3) << 3;
    int gr = m0 + r, gc = k0 + c;
    unsigned short* dp = &sA[buf][r * LDSB + c];
    if (gr < M && gc < K) {                       // K is a multiple of 16 here
      async_copy_b128(lds_off_u32(dp), A + (size_t)gr * K + gc);
    } else {
      *reinterpret_cast<uint4*>(dp) = make_uint4(0u,0u,0u,0u); // ragged: zero
    }
  };
  auto stageB = [&](int buf, int k0, int cid) {
    int r = cid >> 2, c = (cid & 3) << 3;
    int gr = n0 + r, gc = k0 + c;
    unsigned short* dp = &sB[buf][r * LDSB + c];
    if (gr < N && gc < K) {
      async_copy_b128(lds_off_u32(dp), Bw + (size_t)gr * K + gc);
    } else {
      *reinterpret_cast<uint4*>(dp) = make_uint4(0u,0u,0u,0u);
    }
  };

  const int nt = (K + BK - 1) / BK;

  // prologue: stage tile 0 into buffer 0
  stageA(0, 0, tid);
  stageA(0, 0, tid + 256);
  stageB(0, 0, tid);
  wait_asynccnt0();
  __syncthreads();

  int buf = 0;
  for (int t = 0; t < nt; ++t) {
    const int k0 = t * BK;

    // kick off next tile into the other buffer (overlaps with WMMA below)
    if (t + 1 < nt) {
      const int k1 = k0 + BK;
      stageA(buf ^ 1, k1, tid);
      stageA(buf ^ 1, k1, tid + 256);
      stageB(buf ^ 1, k1, tid);
      if (k1 + BK < K)  // pull tile t+2 toward L2 (global_prefetch_b8)
        __builtin_prefetch(A + (size_t)(m0 + (tid >> 1)) * K + (k1 + BK), 0, 1);
    }

    // compute on current buffer
    FragU fa[2], fb[2];
    #pragma unroll
    for (int mi = 0; mi < 2; ++mi) {
      int ar = wm * 32 + mi * 16 + frow;
      fa[mi].u[0] = *reinterpret_cast<const uint4*>(&sA[buf][ar * LDSB + fkb]);
      fa[mi].u[1] = *reinterpret_cast<const uint4*>(&sA[buf][ar * LDSB + fkb + 16]);
    }
    #pragma unroll
    for (int ni = 0; ni < 2; ++ni) {
      int br = wn * 32 + ni * 16 + frow;
      fb[ni].u[0] = *reinterpret_cast<const uint4*>(&sB[buf][br * LDSB + fkb]);
      fb[ni].u[1] = *reinterpret_cast<const uint4*>(&sB[buf][br * LDSB + fkb + 16]);
    }
    #pragma unroll
    for (int mi = 0; mi < 2; ++mi)
      #pragma unroll
      for (int ni = 0; ni < 2; ++ni)
        acc[mi][ni] = __builtin_amdgcn_wmma_f32_16x16x32_bf16(
            false, fa[mi].v, false, fb[ni].v, (short)0, acc[mi][ni], false, false);

    wait_asynccnt0();   // next-tile copies done (overlapped with WMMA above)
    __syncthreads();
    buf ^= 1;
  }

  // store: VGPR v holds M = v (lanes 0-15) or v+8 (lanes 16-31); N = lane&15
  const int rbase = (lane >> 4) << 3;
  const int cn    = lane & 15;
  #pragma unroll
  for (int mi = 0; mi < 2; ++mi) {
    #pragma unroll
    for (int ni = 0; ni < 2; ++ni) {
      int cc = n0 + wn * 32 + ni * 16 + cn;
      if (cc >= N) continue;
      float bv = (EPI >= 1) ? bias[cc] : 0.f;
      #pragma unroll
      for (int v = 0; v < 8; ++v) {
        int cr = m0 + wm * 32 + mi * 16 + rbase + v;
        if (cr >= M) continue;
        float val = acc[mi][ni][v] + bv;
        if (EPI == 2) val = (val > 20.f) ? val : log1pf(__expf(val));
        C[(size_t)cr * N + cc] = val;
      }
    }
  }
}

// ---------------- depthwise causal conv + SiLU ----------------
__global__ void k_conv_silu(const float* __restrict__ xz,  // [T, 2*DI], xc = cols [0,DI)
                            const float* __restrict__ cw,  // [DI, DCONV]
                            const float* __restrict__ cb,  // [DI]
                            float* __restrict__ xc) {      // [T, DI]
  int i = blockIdx.x * blockDim.x + threadIdx.x;
  if (i >= TTOK * DI) return;
  int d = i % DI;
  int t = i / DI;
  int l = t % SEQ_;
  int b = t / SEQ_;
  float acc = cb[d];
  #pragma unroll
  for (int j = 0; j < DCONV_; ++j) {
    int li = l - (DCONV_ - 1) + j;
    if (li >= 0)
      acc += cw[d * DCONV_ + j] * xz[((size_t)b * SEQ_ + li) * (2 * DI) + d];
  }
  xc[i] = acc / (1.f + __expf(-acc));   // silu
}

// ---------------- selective scan: one thread per (b, channel) ----------------
__global__ void __launch_bounds__(256) k_scan(
    const float* __restrict__ dt,    // [T, DI] (already softplus'ed)
    const float* __restrict__ dbc,   // [T, 80] (B at +48, C at +64)
    const float* __restrict__ xc,    // [T, DI]
    const float* __restrict__ xz,    // [T, 2*DI] (z = cols [DI, 2*DI))
    const float* __restrict__ A_log, // [DI, DS]
    const float* __restrict__ Dsk,   // [DI]
    float* __restrict__ y,           // [T, DI]
    float* __restrict__ stateOut)    // [BATCH, DI, DS] (this layer's slot in d_out)
{
  int id = blockIdx.x * blockDim.x + threadIdx.x;
  if (id >= BATCH_ * DI) return;
  int b = id / DI, d = id - b * DI;
  float Ak[DS], h[DS];
  #pragma unroll
  for (int s = 0; s < DS; ++s) { Ak[s] = -__expf(A_log[d * DS + s]); h[s] = 0.f; }
  float dskip = Dsk[d];
  for (int l = 0; l < SEQ_; ++l) {
    size_t idx = (size_t)b * SEQ_ + l;
    float dtv = dt[idx * DI + d];
    float xv  = xc[idx * DI + d];
    float zv  = xz[idx * (2 * DI) + DI + d];
    const float* Bp = dbc + idx * DBCN + DTR;
    const float* Cp = Bp + DS;
    float dtx = dtv * xv;
    float yv = 0.f;
    #pragma unroll
    for (int s = 0; s < DS; ++s) {
      h[s] = __expf(dtv * Ak[s]) * h[s] + dtx * Bp[s];
      yv += h[s] * Cp[s];
    }
    yv += dskip * xv;
    yv *= zv / (1.f + __expf(-zv));   // * silu(z)
    y[idx * DI + d] = yv;
  }
  #pragma unroll
  for (int s = 0; s < DS; ++s) stateOut[(size_t)(b * DI + d) * DS + s] = h[s];
}

// ---------------- tiny AdaLN projection: ss[b,e] = cond[b]·aw[e] + ab[e] ----------------
__global__ void k_adaln(const float* __restrict__ cond, const float* __restrict__ aw,
                        const float* __restrict__ ab, float* __restrict__ ss) {
  int i = blockIdx.x * blockDim.x + threadIdx.x;
  if (i >= BATCH_ * 2 * DM) return;
  int b = i / (2 * DM), e = i - b * (2 * DM);
  float acc = ab[e];
  const float* cp = cond + b * DCOND_;
  const float* wp = aw + (size_t)e * DCOND_;
  #pragma unroll 4
  for (int k = 0; k < DCOND_; ++k) acc += cp[k] * wp[k];
  ss[i] = acc;
}

// ---------------- wave32 block reduction (sum, sumsq) ----------------
__device__ inline float2 block_reduce_pair(float a, float b) {
  __shared__ float sa[16], sb[16];
  int lane = threadIdx.x & 31, w = threadIdx.x >> 5;
  #pragma unroll
  for (int off = 16; off > 0; off >>= 1) {
    a += __shfl_down(a, off, 32);
    b += __shfl_down(b, off, 32);
  }
  __syncthreads();
  if (lane == 0) { sa[w] = a; sb[w] = b; }
  __syncthreads();
  int nw = (blockDim.x + 31) >> 5;
  if (w == 0) {
    a = (lane < nw) ? sa[lane] : 0.f;
    b = (lane < nw) ? sb[lane] : 0.f;
    #pragma unroll
    for (int off = 8; off > 0; off >>= 1) {
      a += __shfl_down(a, off, 32);
      b += __shfl_down(b, off, 32);
    }
    if (lane == 0) { sa[0] = a; sb[0] = b; }
  }
  __syncthreads();
  return make_float2(sa[0], sb[0]);
}

// ---------------- LN(resid+m) then LN_noaffine * (1+scale) + shift ----------------
__global__ void __launch_bounds__(256) k_ln_mod(
    const float* __restrict__ resid, const float* __restrict__ mout,
    const float* __restrict__ nw, const float* __restrict__ nb,
    const float* __restrict__ ss, float* __restrict__ hout)
{
  int row = blockIdx.x;              // 0..T-1
  int b = row / SEQ_;
  const float* r = resid + (size_t)row * DM;
  const float* m = mout  + (size_t)row * DM;
  float v[3], s0 = 0.f, s1 = 0.f;
  #pragma unroll
  for (int j = 0; j < 3; ++j) {
    int c = threadIdx.x + j * 256;
    float x = r[c] + m[c];
    v[j] = x; s0 += x; s1 += x * x;
  }
  float2 t = block_reduce_pair(s0, s1);
  float mu = t.x / DM;
  float rs = rsqrtf(t.y / DM - mu * mu + EPS_);
  float w2[3];
  s0 = 0.f; s1 = 0.f;
  #pragma unroll
  for (int j = 0; j < 3; ++j) {
    int c = threadIdx.x + j * 256;
    float hn = (v[j] - mu) * rs * nw[c] + nb[c];
    w2[j] = hn; s0 += hn; s1 += hn * hn;
  }
  t = block_reduce_pair(s0, s1);
  float mu2 = t.x / DM;
  float rs2 = rsqrtf(t.y / DM - mu2 * mu2 + EPS_);
  #pragma unroll
  for (int j = 0; j < 3; ++j) {
    int c = threadIdx.x + j * 256;
    float hn2 = (w2[j] - mu2) * rs2;
    float sc = ss[(size_t)b * (2 * DM) + c];
    float sh = ss[(size_t)b * (2 * DM) + DM + c];
    hout[(size_t)row * DM + c] = hn2 * (1.f + sc) + sh;
  }
}

// ---------------- final LN into d_out ----------------
__global__ void __launch_bounds__(256) k_ln_final(
    const float* __restrict__ hin, const float* __restrict__ nw,
    const float* __restrict__ nb, float* __restrict__ outp)
{
  int row = blockIdx.x;
  const float* r = hin + (size_t)row * DM;
  float v[3], s0 = 0.f, s1 = 0.f;
  #pragma unroll
  for (int j = 0; j < 3; ++j) {
    int c = threadIdx.x + j * 256;
    float x = r[c];
    v[j] = x; s0 += x; s1 += x * x;
  }
  float2 t = block_reduce_pair(s0, s1);
  float mu = t.x / DM;
  float rs = rsqrtf(t.y / DM - mu * mu + EPS_);
  #pragma unroll
  for (int j = 0; j < 3; ++j) {
    int c = threadIdx.x + j * 256;
    outp[(size_t)row * DM + c] = (v[j] - mu) * rs * nw[c] + nb[c];
  }
}

// ---------------- host orchestration ----------------
extern "C" void kernel_launch(void* const* d_in, const int* in_sizes, int n_in,
                              void* d_out, int out_size, void* d_ws, size_t ws_size,
                              hipStream_t stream) {
  (void)in_sizes; (void)n_in; (void)out_size; (void)ws_size;
  const float* x      = (const float*)d_in[0];
  const float* cond   = (const float*)d_in[1];
  const float* in_w   = (const float*)d_in[2];
  const float* conv_w = (const float*)d_in[3];
  const float* conv_b = (const float*)d_in[4];
  const float* xp_w   = (const float*)d_in[5];
  const float* dtp_w  = (const float*)d_in[6];
  const float* dtp_b  = (const float*)d_in[7];
  const float* A_log  = (const float*)d_in[8];
  const float* D_skip = (const float*)d_in[9];
  const float* out_w  = (const float*)d_in[10];
  const float* norm_w = (const float*)d_in[11];
  const float* norm_b = (const float*)d_in[12];
  const float* ad_w   = (const float*)d_in[13];
  const float* ad_b   = (const float*)d_in[14];
  const float* fn_w   = (const float*)d_in[15];
  const float* fn_b   = (const float*)d_in[16];
  float* out = (float*)d_out;

  char* ws = (char*)d_ws;
  size_t off = 0;
  auto alloc = [&](size_t bytes) -> void* {
    void* p = ws + off;
    off = (off + bytes + 255) & ~(size_t)255;
    return p;
  };
  unsigned short* wInB  = (unsigned short*)alloc((size_t)NL * 2 * DI * DM * 2);
  unsigned short* wXpB  = (unsigned short*)alloc((size_t)NL * DBCN * DI * 2);
  unsigned short* wDtB  = (unsigned short*)alloc((size_t)NL * DI * DTR * 2);
  unsigned short* wOutB = (unsigned short*)alloc((size_t)NL * DM * DI * 2);
  float* hBuf   = (float*)alloc((size_t)TTOK * DM * 4);
  float* mOut   = (float*)alloc((size_t)TTOK * DM * 4);
  unsigned short* aBf = (unsigned short*)alloc((size_t)TTOK * DI * 2);
  float* xzBuf  = (float*)alloc((size_t)TTOK * 2 * DI * 4);
  float* xcBuf  = (float*)alloc((size_t)TTOK * DI * 4);
  float* dbcBuf = (float*)alloc((size_t)TTOK * DBCN * 4);
  float* dtBuf  = (float*)alloc((size_t)TTOK * DI * 4);
  float* yBuf   = (float*)alloc((size_t)TTOK * DI * 4);
  float* ssBuf  = (float*)alloc((size_t)BATCH_ * 2 * DM * 4);

  auto cvt = [&](const float* s, unsigned short* d, int n) {
    k_cvt_bf16<<<(n + 255) / 256, 256, 0, stream>>>(s, d, n);
  };
  // weights -> bf16 (L2-resident working set: ~44 MB of 192 MB L2)
  cvt(in_w,  wInB,  NL * 2 * DI * DM);
  cvt(xp_w,  wXpB,  NL * DBCN * DI);
  cvt(dtp_w, wDtB,  NL * DI * DTR);
  cvt(out_w, wOutB, NL * DM * DI);

  hipMemcpyAsync(hBuf, x, (size_t)TTOK * DM * 4, hipMemcpyDeviceToDevice, stream);

  for (int i = 0; i < NL; ++i) {
    // in_proj: xz = h * Win^T  [2048 x 3072]
    cvt(hBuf, aBf, TTOK * DM);
    dim3 g1(2 * DI / BN, TTOK / BM);
    k_gemm_wmma<0><<<g1, 256, 0, stream>>>(aBf, wInB + (size_t)i * 2 * DI * DM,
                                           nullptr, xzBuf, TTOK, 2 * DI, DM);
    // depthwise conv + silu
    k_conv_silu<<<(TTOK * DI + 255) / 256, 256, 0, stream>>>(
        xzBuf, conv_w + (size_t)i * DI * DCONV_, conv_b + (size_t)i * DI, xcBuf);
    // x_proj: dbc = xc * Wxp^T  [2048 x 80]
    cvt(xcBuf, aBf, TTOK * DI);
    dim3 g2((DBCN + BN - 1) / BN, TTOK / BM);
    k_gemm_wmma<0><<<g2, 256, 0, stream>>>(aBf, wXpB + (size_t)i * DBCN * DI,
                                           nullptr, dbcBuf, TTOK, DBCN, DI);
    // dt_proj + softplus: dt = softplus(dbc[:, :48] * Wdt^T + b)  [2048 x 1536]
    int ndt = TTOK * DTR;
    k_cvt_bf16_2d<<<(ndt + 255) / 256, 256, 0, stream>>>(dbcBuf, aBf, TTOK, DTR, DBCN);
    dim3 g3(DI / BN, TTOK / BM);
    k_gemm_wmma<2><<<g3, 256, 0, stream>>>(aBf, wDtB + (size_t)i * DI * DTR,
                                           dtp_b + (size_t)i * DI, dtBuf, TTOK, DI, DTR);
    // selective scan (fused D-skip + silu(z) gate); final states -> d_out tail
    float* stOut = out + (size_t)TTOK * DM + (size_t)i * BATCH_ * DI * DS;
    k_scan<<<(BATCH_ * DI + 255) / 256, 256, 0, stream>>>(
        dtBuf, dbcBuf, xcBuf, xzBuf,
        A_log + (size_t)i * DI * DS, D_skip + (size_t)i * DI, yBuf, stOut);
    // out_proj: m = y * Wout^T  [2048 x 768]
    cvt(yBuf, aBf, TTOK * DI);
    dim3 g4(DM / BN, TTOK / BM);
    k_gemm_wmma<0><<<g4, 256, 0, stream>>>(aBf, wOutB + (size_t)i * DM * DI,
                                           nullptr, mOut, TTOK, DM, DI);
    // adaln scale/shift
    k_adaln<<<(BATCH_ * 2 * DM + 255) / 256, 256, 0, stream>>>(
        cond, ad_w + (size_t)i * 2 * DM * DCOND_, ad_b + (size_t)i * 2 * DM, ssBuf);
    // LN + modulated LN (in-place on h, one block per row)
    k_ln_mod<<<TTOK, 256, 0, stream>>>(hBuf, mOut, norm_w + (size_t)i * DM,
                                       norm_b + (size_t)i * DM, ssBuf, hBuf);
  }
  k_ln_final<<<TTOK, 256, 0, stream>>>(hBuf, fn_w, fn_b, out);
}
